// eComformer_76940044141022
// MI455X (gfx1250) — matrix operand order repaired and checked by reading, not compiled, not run on hardware
//
#include <hip/hip_runtime.h>
#include <hip/hip_bf16.h>
#include <math.h>

// ---------------------------------------------------------------------------
// Problem constants (from the reference)
// ---------------------------------------------------------------------------
#define N_NODES 2048
#define N_EDGES 16384
#define CDIM    256
#define NGRAPH  64
#define NSD     64      // NS
#define NVD     8       // NV
#define WLEN    5120    // WN1 == WN2 == 64*64 + 64*8 + 64*8
#define ECHUNK  2048    // edge chunk for the 5120-wide filter GEMMs

#define ACT_NONE     0
#define ACT_SOFTPLUS 1
#define ACT_RELU     2
#define ACT_SILU     3

#define SQ3F  1.7320508075688772f
#define SQ5F  2.23606797749979f
#define SQ15F 3.872983346207417f

typedef __attribute__((ext_vector_type(16))) __bf16 bf16x16;
typedef __attribute__((ext_vector_type(8)))  float  f32x8;

__device__ __forceinline__ float act_apply(float x, int act) {
    switch (act) {
        case ACT_SOFTPLUS: return (x > 20.0f) ? x : log1pf(expf(x));
        case ACT_RELU:     return fmaxf(x, 0.0f);
        case ACT_SILU:     return x / (1.0f + expf(-x));
        default:           return x;
    }
}

// ---------------------------------------------------------------------------
// Weight pre-pack: W[K,N] (f32, row-major) -> bf16 in WMMA B-fragment order.
// Tile (nt, ks) holds 512 bf16: lane 0..31, elem 0..15 with
//   n = nt*16 + (lane&15),  k = ks*32 + (lane>>4)*8 + (e&7) + (e>>3)*16   (zero padded OOB)
// ---------------------------------------------------------------------------
__global__ void k_pack(const float* __restrict__ W, __bf16* __restrict__ P,
                       int K, int N, int kSteps, int total)
{
    int t = blockIdx.x * blockDim.x + threadIdx.x;
    if (t >= total) return;
    int e    = t & 15;
    int lane = (t >> 4) & 31;
    int ks   = (t >> 9) % kSteps;
    int nt   = t / (kSteps << 9);
    int n = nt * 16 + (lane & 15);
    int k = ks * 32 + (lane >> 4) * 8 + (e & 7) + (e >> 3) * 16;
    float v = (k < K && n < N) ? W[(size_t)k * N + n] : 0.0f;
    P[t] = (__bf16)v;
}

// Zero-padded copy of x (2048 x 92) into (2048 x 96)
__global__ void k_padcopy(const float* __restrict__ in, float* __restrict__ out, int total)
{
    int t = blockIdx.x * blockDim.x + threadIdx.x;
    if (t >= total) return;
    int r = t / 92, c = t % 92;
    out[(size_t)r * 96 + c] = in[t];
}

// ---------------------------------------------------------------------------
// GEMM: C[M,N] = act(A[M,Kpad] @ W + bias), W pre-packed bf16 fragments.
// One wave computes a 32(M) x 64(N) tile: 2 m-subtiles x 4 n-subtiles,
// 8 f32x8 accumulators. Per k-step in the full path: 16 float2 A loads +
// 4x 32B B loads issued as one clause + 8 back-to-back WMMAs (A frags
// reused 4x, B frags 2x).  Requires M % 32 == 0 (true at every call site).
// ---------------------------------------------------------------------------
__global__ void __launch_bounds__(32)
k_gemm(const float* __restrict__ A, const __bf16* __restrict__ P,
       const float* __restrict__ bias, float* __restrict__ C,
       int M, int N, int Kpad, int kSteps, int nTilesPad, int act)
{
    const int lane = threadIdx.x & 31;
    const int nt0  = blockIdx.x * 4;
    const int m0   = blockIdx.y * 32;

    const int h  = lane >> 4;             // half-wave select
    const int ml = lane & 15;

    const float*  Arow0 = A + (size_t)(m0 + ml) * Kpad + 8 * h;
    const float*  Arow1 = Arow0 + (size_t)16 * Kpad;
    const __bf16* Pl    = P + lane * 16;

    f32x8 acc[2][4];
    #pragma unroll
    for (int i = 0; i < 2; ++i)
        #pragma unroll
        for (int j = 0; j < 4; ++j)
            acc[i][j] = (f32x8){0.f, 0.f, 0.f, 0.f, 0.f, 0.f, 0.f, 0.f};

    if (nt0 + 4 <= nTilesPad) {
        // ---- full path: no guards inside the K loop ----
        for (int ks = 0; ks < kSteps; ++ks) {
            const float* ap0 = Arow0 + ks * 32;
            const float* ap1 = Arow1 + ks * 32;
            if (ks + 1 < kSteps) {                // hide the f32 A stream
                __builtin_prefetch(ap0 + 32, 0, 3);
                __builtin_prefetch(ap1 + 32, 0, 3);
            }
            bf16x16 a0, a1;
            #pragma unroll
            for (int r = 0; r < 8; ++r) {
                const int kb = ((r < 4) ? 0 : 16) + 2 * (r & 3);
                float2 v0 = *(const float2*)(ap0 + kb);
                float2 v1 = *(const float2*)(ap1 + kb);
                a0[2 * r + 0] = (__bf16)v0.x;  a0[2 * r + 1] = (__bf16)v0.y;
                a1[2 * r + 0] = (__bf16)v1.x;  a1[2 * r + 1] = (__bf16)v1.y;
            }
            // issue all four B-fragment loads first (one clause, staggered waits)
            bf16x16 b[4];
            #pragma unroll
            for (int j = 0; j < 4; ++j)
                b[j] = *(const bf16x16*)(Pl + ((((size_t)(nt0 + j) * kSteps + ks) << 9)));
            #pragma unroll
            for (int j = 0; j < 4; ++j) {
                acc[0][j] = __builtin_amdgcn_wmma_f32_16x16x32_bf16(false, a0, false, b[j],
                                                                    (short)0, acc[0][j], false, false);
                acc[1][j] = __builtin_amdgcn_wmma_f32_16x16x32_bf16(false, a1, false, b[j],
                                                                    (short)0, acc[1][j], false, false);
            }
        }
    } else {
        // ---- tail path (only the N=1 head GEMM lands here) ----
        for (int ks = 0; ks < kSteps; ++ks) {
            const float* ap0 = Arow0 + ks * 32;
            const float* ap1 = Arow1 + ks * 32;
            bf16x16 a0, a1;
            #pragma unroll
            for (int r = 0; r < 8; ++r) {
                const int kb = ((r < 4) ? 0 : 16) + 2 * (r & 3);
                float2 v0 = *(const float2*)(ap0 + kb);
                float2 v1 = *(const float2*)(ap1 + kb);
                a0[2 * r + 0] = (__bf16)v0.x;  a0[2 * r + 1] = (__bf16)v0.y;
                a1[2 * r + 0] = (__bf16)v1.x;  a1[2 * r + 1] = (__bf16)v1.y;
            }
            #pragma unroll
            for (int j = 0; j < 4; ++j) {
                if (nt0 + j < nTilesPad) {        // uniform guard, EXEC untouched
                    bf16x16 b = *(const bf16x16*)(Pl + ((((size_t)(nt0 + j) * kSteps + ks) << 9)));
                    acc[0][j] = __builtin_amdgcn_wmma_f32_16x16x32_bf16(false, a0, false, b,
                                                                        (short)0, acc[0][j], false, false);
                    acc[1][j] = __builtin_amdgcn_wmma_f32_16x16x32_bf16(false, a1, false, b,
                                                                        (short)0, acc[1][j], false, false);
                }
            }
        }
    }

    // D 16x16 f32: lane = column, VGPR v = row v (+8 for upper half)
    #pragma unroll
    for (int j = 0; j < 4; ++j) {
        const int n = (nt0 + j) * 16 + ml;
        if (n < N) {
            const float bv = bias[n];
            #pragma unroll
            for (int i = 0; i < 2; ++i) {
                #pragma unroll
                for (int v = 0; v < 8; ++v) {
                    const int m = m0 + i * 16 + v + 8 * h;   // M % 32 == 0
                    C[(size_t)m * N + n] = act_apply(acc[i][j][v] + bv, act);
                }
            }
        }
    }
}

// ---------------------------------------------------------------------------
// Edge RBF features: d = -0.75/|vec|; rbf = exp(-gamma*(d-center_c)^2)
// ---------------------------------------------------------------------------
__global__ void k_rbf(const float* __restrict__ ea, float* __restrict__ out, int total)
{
    int t = blockIdx.x * blockDim.x + threadIdx.x;
    if (t >= total) return;
    int e = t >> 8, c = t & 255;
    float x = ea[e * 3 + 0], y = ea[e * 3 + 1], z = ea[e * 3 + 2];
    float d = -0.75f * rsqrtf(x * x + y * y + z * z);
    float center = -4.0f + 4.0f * (float)c / 255.0f;
    float gamma  = 255.0f / 4.0f;
    float diff = d - center;
    out[t] = expf(-gamma * diff * diff);
}

// Gather+concat: out[e] = [nodeA[idxA[e]], nodeB[idxB[e]], edgeE[e]]  (3*256 wide)
__global__ void k_gcat(const float* __restrict__ nodeA, const int* __restrict__ idxA,
                       const float* __restrict__ nodeB, const int* __restrict__ idxB,
                       const float* __restrict__ edgeE, float* __restrict__ out, int total)
{
    int t = blockIdx.x * blockDim.x + threadIdx.x;
    if (t >= total) return;
    int e = t / (3 * CDIM), c = t % (3 * CDIM);
    float v;
    if (c < CDIM)          v = nodeA[(size_t)idxA[e] * CDIM + c];
    else if (c < 2 * CDIM) v = nodeB[(size_t)idxB[e] * CDIM + (c - CDIM)];
    else                   v = edgeE[(size_t)e * CDIM + (c - 2 * CDIM)];
    out[t] = v;
}

// alpha[e,c] = q[dst[e],c] * kj[e,c] / sqrt(C)
__global__ void k_alpha(const float* __restrict__ q, const int* __restrict__ dst,
                        const float* __restrict__ kj, float* __restrict__ alpha, int total)
{
    int t = blockIdx.x * blockDim.x + threadIdx.x;
    if (t >= total) return;
    int e = t >> 8, c = t & 255;
    alpha[t] = q[(size_t)dst[e] * CDIM + c] * kj[t] * 0.0625f;  // 1/sqrt(256)
}

// Row layernorm over C=256 (one block per row)
__global__ void k_layernorm(const float* __restrict__ in, const float* __restrict__ g,
                            const float* __restrict__ b, float* __restrict__ out, int rows)
{
    int r = blockIdx.x, c = threadIdx.x;
    __shared__ float red[256];
    __shared__ float mu_s, var_s;
    float x = in[(size_t)r * CDIM + c];
    red[c] = x; __syncthreads();
    for (int d = 128; d > 0; d >>= 1) { if (c < d) red[c] += red[c + d]; __syncthreads(); }
    if (c == 0) mu_s = red[0] * (1.0f / 256.0f);
    __syncthreads();
    float dv = x - mu_s;
    red[c] = dv * dv; __syncthreads();
    for (int d = 128; d > 0; d >>= 1) { if (c < d) red[c] += red[c + d]; __syncthreads(); }
    if (c == 0) var_s = red[0] * (1.0f / 256.0f);
    __syncthreads();
    out[(size_t)r * CDIM + c] = dv * rsqrtf(var_s + 1e-5f) * g[c] + b[c];
}

// out = msg * sigmoid(ln)
__global__ void k_sigmul(const float* __restrict__ msg, const float* __restrict__ ln,
                         float* __restrict__ out, int total)
{
    int t = blockIdx.x * blockDim.x + threadIdx.x;
    if (t >= total) return;
    out[t] = msg[t] * (1.0f / (1.0f + expf(-ln[t])));
}

// Scatter add rows: out[idx[e], :D] += in[e, :D]
__global__ void k_scatter_add(const float* __restrict__ in, const int* __restrict__ idx,
                              float* __restrict__ out, int total, int D)
{
    int t = blockIdx.x * blockDim.x + threadIdx.x;
    if (t >= total) return;
    int e = t / D, c = t % D;
    atomicAdd(&out[(size_t)idx[e] * D + c], in[t]);
}

__global__ void k_counts(const int* __restrict__ idx, float* __restrict__ cnt, int n)
{
    int t = blockIdx.x * blockDim.x + threadIdx.x;
    if (t >= n) return;
    atomicAdd(&cnt[idx[t]], 1.0f);
}

// buf[r, :D] /= max(cnt[r], 1)
__global__ void k_divrows(float* __restrict__ buf, const float* __restrict__ cnt,
                          int total, int D)
{
    int t = blockIdx.x * blockDim.x + threadIdx.x;
    if (t >= total) return;
    int r = t / D;
    buf[t] = buf[t] / fmaxf(cnt[r], 1.0f);
}

// Column batchnorm stats over `rows` rows, C=256 columns (one block per column)
__global__ void k_bn_stats(const float* __restrict__ X, int rows,
                           float* __restrict__ mu, float* __restrict__ var)
{
    int col = blockIdx.x, t = threadIdx.x;
    float s = 0.f, ss = 0.f;
    for (int r = t; r < rows; r += 256) {
        float v = X[(size_t)r * CDIM + col];
        s += v; ss += v * v;
    }
    __shared__ float rs[256], rss[256];
    rs[t] = s; rss[t] = ss; __syncthreads();
    for (int d = 128; d > 0; d >>= 1) {
        if (t < d) { rs[t] += rs[t + d]; rss[t] += rss[t + d]; }
        __syncthreads();
    }
    if (t == 0) {
        float m = rs[0] / (float)rows;
        mu[col]  = m;
        var[col] = rss[0] / (float)rows - m * m;
    }
}

// h = softplus(h + (y - mu)*rsqrt(var+eps)*g + b)   (in place on h)
__global__ void k_bn_res_softplus(float* __restrict__ h, const float* __restrict__ y,
                                  const float* __restrict__ mu, const float* __restrict__ var,
                                  const float* __restrict__ g, const float* __restrict__ b,
                                  int total)
{
    int t = blockIdx.x * blockDim.x + threadIdx.x;
    if (t >= total) return;
    int c = t & 255;
    float v = h[t] + (y[t] - mu[c]) * rsqrtf(var[c] + 1e-5f) * g[c] + b[c];
    h[t] = (v > 20.0f) ? v : log1pf(expf(v));
}

// Spherical harmonics of normalized edge vectors
__global__ void k_sph(const float* __restrict__ ea, float* __restrict__ sh1,
                      float* __restrict__ sh2, int E)
{
    int e = blockIdx.x * blockDim.x + threadIdx.x;
    if (e >= E) return;
    float x = ea[e * 3 + 0], y = ea[e * 3 + 1], z = ea[e * 3 + 2];
    float inv = rsqrtf(x * x + y * y + z * z);
    x *= inv; y *= inv; z *= inv;
    sh1[e * 3 + 0] = SQ3F * y;
    sh1[e * 3 + 1] = SQ3F * z;
    sh1[e * 3 + 2] = SQ3F * x;
    sh2[e * 5 + 0] = SQ15F * x * y;
    sh2[e * 5 + 1] = SQ15F * y * z;
    sh2[e * 5 + 2] = 0.5f * SQ5F * (2.f * z * z - x * x - y * y);
    sh2[e * 5 + 3] = SQ15F * x * z;
    sh2[e * 5 + 4] = 0.5f * SQ15F * (x * x - y * y);
}

// equi phase 1: per edge, t0/t1/t2 einsums from filter row wv[5120], scatter into n0/n1/n2
__global__ void k_equi1(const float* __restrict__ wbuf, const float* __restrict__ hemb,
                        const int* __restrict__ src, const int* __restrict__ dst,
                        const float* __restrict__ sh1, const float* __restrict__ sh2,
                        float* __restrict__ n0, float* __restrict__ n1,
                        float* __restrict__ n2, int ebase)
{
    int er = blockIdx.x;
    int e  = ebase + er;
    int w  = threadIdx.x;                      // 0..63
    __shared__ float xj[NSD];
    int d = dst[e];
    xj[w] = hemb[(size_t)d * NSD + w];
    __syncthreads();
    const float* wv = wbuf + (size_t)er * WLEN;
    const float c1 = 0.125f;                   // 1/sqrt(64)
    float t0 = 0.f;
    for (int u = 0; u < NSD; ++u) t0 += xj[u] * wv[u * NSD + w];
    t0 *= c1;
    int s = src[e];
    atomicAdd(&n0[(size_t)s * NSD + w], t0);
    if (w < NVD) {
        float a1 = 0.f, a2 = 0.f;
        for (int u = 0; u < NSD; ++u) {
            a1 += xj[u] * wv[4096 + u * NVD + w];
            a2 += xj[u] * wv[4608 + u * NVD + w];
        }
        a1 *= c1; a2 *= c1;
        for (int m = 0; m < 3; ++m)
            atomicAdd(&n1[(size_t)s * (NVD * 3) + w * 3 + m], a1 * sh1[e * 3 + m]);
        for (int m = 0; m < 5; ++m)
            atomicAdd(&n2[(size_t)s * (NVD * 5) + w * 5 + m], a2 * sh2[e * 5 + m]);
    }
}

// equi phase 2: per edge, s0+s1+s2 from second filter row, scatter into gsum
__global__ void k_equi2(const float* __restrict__ wbuf, const float* __restrict__ n0,
                        const float* __restrict__ n1, const float* __restrict__ n2,
                        const int* __restrict__ src, const int* __restrict__ dst,
                        const float* __restrict__ sh1, const float* __restrict__ sh2,
                        float* __restrict__ gsum, int ebase)
{
    int er = blockIdx.x;
    int e  = ebase + er;
    int w  = threadIdx.x;                      // 0..63
    __shared__ float n0d[NSD], p1[NVD], p2[NVD];
    int d = dst[e];
    n0d[w] = n0[(size_t)d * NSD + w];
    if (w < NVD) {
        float a = 0.f;
        for (int m = 0; m < 3; ++m) a += n1[(size_t)d * (NVD * 3) + w * 3 + m] * sh1[e * 3 + m];
        p1[w] = a / SQ3F;
        float b = 0.f;
        for (int m = 0; m < 5; ++m) b += n2[(size_t)d * (NVD * 5) + w * 5 + m] * sh2[e * 5 + m];
        p2[w] = b / SQ5F;
    }
    __syncthreads();
    const float* wv = wbuf + (size_t)er * WLEN;
    float s0 = 0.f;
    for (int u = 0; u < NSD; ++u) s0 += wv[u * NSD + w] * n0d[u];
    float s1 = 0.f, s2 = 0.f;
    for (int v = 0; v < NVD; ++v) {
        s1 += wv[4096 + v * NSD + w] * p1[v];
        s2 += wv[4608 + v * NSD + w] * p2[v];
    }
    const float pw = 0.11180339887498948f;     // 1/sqrt(64+8+8)
    atomicAdd(&gsum[(size_t)src[e] * NSD + w], pw * (s0 + s1 + s2));
}

// ---------------------------------------------------------------------------
// Host side
// ---------------------------------------------------------------------------
static inline int divup(int a, int b) { return (a + b - 1) / b; }

struct AttP {
    const float *qb,*kb,*vb,*eb,*ku1b,*ku2b,*mu1b,*mu2b,*msgb;
    const float *lnmg,*lnmb,*lnag,*lnab,*catb,*bng,*bnb;
    const __bf16 *Pq,*Pk,*Pv,*Pe,*Pku1,*Pku2,*Pmu1,*Pmu2,*Pmsg,*Pcat;
};

extern "C" void kernel_launch(void* const* d_in, const int* in_sizes, int n_in,
                              void* d_out, int out_size, void* d_ws, size_t ws_size,
                              hipStream_t stream)
{
    (void)in_sizes; (void)n_in; (void)out_size; (void)ws_size;

    // --- input walk (setup_inputs() insertion order; each _lin is {w, b}) ---
    int pi = 0;
    auto nf = [&]() { return (const float*)d_in[pi++]; };
    const float* x         = nf();   // (2048, 92)
    const float* edge_attr = nf();   // (16384, 3)
    const float* aew = nf(); const float* aeb = nf();   // atom_emb
    const float* rlw = nf(); const float* rlb = nf();   // rbf_lin
    const float* attw[3][10];
    AttP att[3];
    for (int a = 0; a < 3; ++a) {
        AttP& P = att[a];
        const float* qw  = nf(); P.qb   = nf();
        const float* kw  = nf(); P.kb   = nf();
        const float* vw  = nf(); P.vb   = nf();
        const float* ew  = nf(); P.eb   = nf();
        const float* ku1w= nf(); P.ku1b = nf();
        const float* ku2w= nf(); P.ku2b = nf();
        const float* mu1w= nf(); P.mu1b = nf();
        const float* mu2w= nf(); P.mu2b = nf();
        const float* msgw= nf(); P.msgb = nf();
        P.lnmg = nf(); P.lnmb = nf(); P.lnag = nf(); P.lnab = nf();
        const float* catw= nf(); P.catb = nf();
        P.bng = nf(); P.bnb = nf();
        attw[a][0]=qw; attw[a][1]=kw; attw[a][2]=vw; attw[a][3]=ew;
        attw[a][4]=ku1w; attw[a][5]=ku2w; attw[a][6]=mu1w; attw[a][7]=mu2w;
        attw[a][8]=msgw; attw[a][9]=catw;
    }
    const float* embw  = nf(); const float* embb  = nf();
    const float* fc1aw = nf(); const float* fc1ab = nf();
    const float* fc1bw = nf(); const float* fc1bb = nf();
    const float* fc2aw = nf(); const float* fc2ab = nf();
    const float* fc2bw = nf(); const float* fc2bb = nf();
    const float* upw   = nf(); const float* upb   = nf();
    const float* ebng  = nf(); const float* ebnb  = nf();
    const float* fcw   = nf(); const float* fcb   = nf();
    const float* fow   = nf(); const float* fob   = nf();
    const int* edge_index = (const int*)d_in[pi++];  // (2, 16384)
    const int* batch      = (const int*)d_in[pi++];  // (2048,)
    const int* src = edge_index;                     // ei[0] == j
    const int* dst = edge_index + N_EDGES;           // ei[1] == i

    // --- workspace arena (64B-aligned byte allocator) ---
    char* wsb = (char*)d_ws;
    size_t off = 0;
    auto wsa = [&](size_t bytes) {
        off = (off + 63) & ~(size_t)63;
        void* p = wsb + off;
        off += bytes;
        return p;
    };
    auto fbuf = [&](size_t n) { return (float*)wsa(n * sizeof(float)); };

    float* h     = fbuf((size_t)N_NODES * CDIM);
    float* ea    = fbuf((size_t)N_EDGES * CDIM);
    float* nq    = fbuf((size_t)N_NODES * CDIM);
    float* nk    = fbuf((size_t)N_NODES * CDIM);
    float* nv    = fbuf((size_t)N_NODES * CDIM);
    float* eLin  = fbuf((size_t)N_EDGES * CDIM);
    float* cat   = fbuf((size_t)N_EDGES * 3 * CDIM);  // also holds RBF early
    float* hid   = fbuf((size_t)N_EDGES * CDIM);
    float* kj    = fbuf((size_t)N_EDGES * CDIM);
    float* alpha = fbuf((size_t)N_EDGES * CDIM);
    float* msg   = fbuf((size_t)N_EDGES * CDIM);
    float* lnA   = fbuf((size_t)N_EDGES * CDIM);
    float* oute  = fbuf((size_t)N_EDGES * CDIM);
    float* agg   = fbuf((size_t)N_NODES * CDIM);
    float* outb  = fbuf((size_t)N_NODES * CDIM);
    float* mu    = fbuf(CDIM);
    float* var   = fbuf(CDIM);
    float* hemb  = fbuf((size_t)N_NODES * NSD);
    float* wbuf  = fbuf((size_t)ECHUNK * WLEN);
    float* n0    = fbuf((size_t)N_NODES * NSD);
    float* n1    = fbuf((size_t)N_NODES * NVD * 3);
    float* n2    = fbuf((size_t)N_NODES * NVD * 5);
    float* cnts  = fbuf(N_NODES);
    float* gsum  = fbuf((size_t)N_NODES * NSD);
    float* sh1   = fbuf((size_t)N_EDGES * 3);
    float* sh2   = fbuf((size_t)N_EDGES * 5);
    float* gpool = fbuf((size_t)NGRAPH * CDIM);
    float* gcnt  = fbuf(NGRAPH);
    float* g2    = fbuf((size_t)NGRAPH * CDIM);
    float* xpad  = fbuf((size_t)N_NODES * 96);

    const int TB = 256;
    const int EC = N_EDGES * CDIM;
    const int NC = N_NODES * CDIM;

    // --- pack all weights into WMMA B-fragment bf16 layout ---
    auto pack = [&](const float* W, int K, int N) {
        int kSteps = divup(K, 32);
        int nt     = divup(N, 16);
        int total  = nt * kSteps * 512;
        __bf16* P = (__bf16*)wsa((size_t)total * sizeof(__bf16));
        k_pack<<<divup(total, TB), TB, 0, stream>>>(W, P, K, N, kSteps, total);
        return (const __bf16*)P;
    };
    const __bf16* Paew = pack(aew, 92, CDIM);
    const __bf16* Prl  = pack(rlw, CDIM, CDIM);
    for (int a = 0; a < 3; ++a) {
        att[a].Pq   = pack(attw[a][0], CDIM, CDIM);
        att[a].Pk   = pack(attw[a][1], CDIM, CDIM);
        att[a].Pv   = pack(attw[a][2], CDIM, CDIM);
        att[a].Pe   = pack(attw[a][3], CDIM, CDIM);
        att[a].Pku1 = pack(attw[a][4], 3 * CDIM, CDIM);
        att[a].Pku2 = pack(attw[a][5], CDIM, CDIM);
        att[a].Pmu1 = pack(attw[a][6], 3 * CDIM, CDIM);
        att[a].Pmu2 = pack(attw[a][7], CDIM, CDIM);
        att[a].Pmsg = pack(attw[a][8], CDIM, CDIM);
        att[a].Pcat = pack(attw[a][9], CDIM, CDIM);
    }
    const __bf16* Pemb  = pack(embw,  CDIM, NSD);
    const __bf16* Pfc1a = pack(fc1aw, CDIM, CDIM);
    const __bf16* Pfc1b = pack(fc1bw, CDIM, WLEN);
    const __bf16* Pfc2a = pack(fc2aw, CDIM, CDIM);
    const __bf16* Pfc2b = pack(fc2bw, CDIM, WLEN);
    const __bf16* Pup   = pack(upw,   NSD,  CDIM);
    const __bf16* Pfc   = pack(fcw,   CDIM, CDIM);
    const __bf16* Pfo   = pack(fow,   CDIM, 1);

    // GEMM launcher (M % 32 == 0 at every call site)
    auto gemm = [&](const float* A, const __bf16* P, const float* bias, float* C,
                    int M, int N, int Kpad, int K, int act) {
        int kSteps = divup(K, 32);
        int ntPad  = divup(N, 16);
        dim3 grid(divup(ntPad, 4), M / 32);
        k_gemm<<<grid, 32, 0, stream>>>(A, P, bias, C, M, N, Kpad, kSteps, ntPad, act);
    };

    // --- h = linear(x, atom_emb)  (x zero-padded 92 -> 96) ---
    hipMemsetAsync(xpad, 0, (size_t)N_NODES * 96 * sizeof(float), stream);
    k_padcopy<<<divup(N_NODES * 92, TB), TB, 0, stream>>>(x, xpad, N_NODES * 92);
    gemm(xpad, Paew, aeb, h, N_NODES, CDIM, 96, 92, ACT_NONE);

    // --- ea = softplus(linear(rbf, rbf_lin)) ---
    k_rbf<<<divup(EC, TB), TB, 0, stream>>>(edge_attr, cat, EC);
    gemm(cat, Prl, rlb, ea, N_EDGES, CDIM, CDIM, CDIM, ACT_SOFTPLUS);

    // --- attention conv (shared routine) ---
    auto run_att = [&](const AttP& P) {
        gemm(h,  P.Pq, P.qb, nq,   N_NODES, CDIM, CDIM, CDIM, ACT_NONE);
        gemm(h,  P.Pk, P.kb, nk,   N_NODES, CDIM, CDIM, CDIM, ACT_NONE);
        gemm(h,  P.Pv, P.vb, nv,   N_NODES, CDIM, CDIM, CDIM, ACT_NONE);
        gemm(ea, P.Pe, P.eb, eLin, N_EDGES, CDIM, CDIM, CDIM, ACT_NONE);
        // kj = linear(silu(linear([k[i],k[j],e], ku1)), ku2)
        k_gcat<<<divup(3 * EC, TB), TB, 0, stream>>>(nk, dst, nk, src, eLin, cat, 3 * EC);
        gemm(cat, P.Pku1, P.ku1b, hid, N_EDGES, CDIM, 3 * CDIM, 3 * CDIM, ACT_SILU);
        gemm(hid, P.Pku2, P.ku2b, kj,  N_EDGES, CDIM, CDIM, CDIM, ACT_NONE);
        k_alpha<<<divup(EC, TB), TB, 0, stream>>>(nq, dst, kj, alpha, EC);
        // msg = linear(silu(linear([v[i],v[j],e], mu1)), mu2)
        k_gcat<<<divup(3 * EC, TB), TB, 0, stream>>>(nv, dst, nv, src, eLin, cat, 3 * EC);
        gemm(cat, P.Pmu1, P.mu1b, hid, N_EDGES, CDIM, 3 * CDIM, 3 * CDIM, ACT_SILU);
        gemm(hid, P.Pmu2, P.mu2b, msg, N_EDGES, CDIM, CDIM, CDIM, ACT_NONE);
        // out_e = layernorm(linear(msg * sigmoid(layernorm(alpha)), msg_p))
        k_layernorm<<<N_EDGES, 256, 0, stream>>>(alpha, P.lnag, P.lnab, lnA, N_EDGES);
        k_sigmul<<<divup(EC, TB), TB, 0, stream>>>(msg, lnA, oute, EC);
        gemm(oute, P.Pmsg, P.msgb, alpha, N_EDGES, CDIM, CDIM, CDIM, ACT_NONE);
        k_layernorm<<<N_EDGES, 256, 0, stream>>>(alpha, P.lnmg, P.lnmb, oute, N_EDGES);
        // agg = segment_sum(out_e, i);  out = linear(agg, cat)
        hipMemsetAsync(agg, 0, (size_t)NC * sizeof(float), stream);
        k_scatter_add<<<divup(EC, TB), TB, 0, stream>>>(oute, dst, agg, EC, CDIM);
        gemm(agg, P.Pcat, P.catb, outb, N_NODES, CDIM, CDIM, CDIM, ACT_NONE);
        // h = softplus(h + batchnorm(out))
        k_bn_stats<<<CDIM, 256, 0, stream>>>(outb, N_NODES, mu, var);
        k_bn_res_softplus<<<divup(NC, TB), TB, 0, stream>>>(h, outb, mu, var, P.bng, P.bnb, NC);
    };

    run_att(att[0]);

    // --- equi_update ---
    k_sph<<<divup(N_EDGES, TB), TB, 0, stream>>>(edge_attr, sh1, sh2, N_EDGES);
    gemm(h, Pemb, embb, hemb, N_NODES, NSD, CDIM, CDIM, ACT_NONE);
    hipMemsetAsync(n0,   0, (size_t)N_NODES * NSD     * sizeof(float), stream);
    hipMemsetAsync(n1,   0, (size_t)N_NODES * NVD * 3 * sizeof(float), stream);
    hipMemsetAsync(n2,   0, (size_t)N_NODES * NVD * 5 * sizeof(float), stream);
    hipMemsetAsync(cnts, 0, (size_t)N_NODES           * sizeof(float), stream);
    hipMemsetAsync(gsum, 0, (size_t)N_NODES * NSD     * sizeof(float), stream);
    k_counts<<<divup(N_EDGES, TB), TB, 0, stream>>>(src, cnts, N_EDGES);
    // filter net 1, chunked over edges
    gemm(ea, Pfc1a, fc1ab, hid, N_EDGES, CDIM, CDIM, CDIM, ACT_RELU);
    for (int c = 0; c < N_EDGES / ECHUNK; ++c) {
        gemm(hid + (size_t)c * ECHUNK * CDIM, Pfc1b, fc1bb, wbuf,
             ECHUNK, WLEN, CDIM, CDIM, ACT_NONE);
        k_equi1<<<ECHUNK, NSD, 0, stream>>>(wbuf, hemb, src, dst, sh1, sh2,
                                            n0, n1, n2, c * ECHUNK);
    }
    k_divrows<<<divup(N_NODES * NSD,     TB), TB, 0, stream>>>(n0, cnts, N_NODES * NSD,     NSD);
    k_divrows<<<divup(N_NODES * NVD * 3, TB), TB, 0, stream>>>(n1, cnts, N_NODES * NVD * 3, NVD * 3);
    k_divrows<<<divup(N_NODES * NVD * 5, TB), TB, 0, stream>>>(n2, cnts, N_NODES * NVD * 5, NVD * 5);
    // filter net 2, chunked over edges
    gemm(ea, Pfc2a, fc2ab, hid, N_EDGES, CDIM, CDIM, CDIM, ACT_RELU);
    for (int c = 0; c < N_EDGES / ECHUNK; ++c) {
        gemm(hid + (size_t)c * ECHUNK * CDIM, Pfc2b, fc2bb, wbuf,
             ECHUNK, WLEN, CDIM, CDIM, ACT_NONE);
        k_equi2<<<ECHUNK, NSD, 0, stream>>>(wbuf, n0, n1, n2, src, dst, sh1, sh2,
                                            gsum, c * ECHUNK);
    }
    k_divrows<<<divup(N_NODES * NSD, TB), TB, 0, stream>>>(gsum, cnts, N_NODES * NSD, NSD);
    // up + batchnorm + residual softplus
    gemm(gsum, Pup, upb, outb, N_NODES, CDIM, NSD, NSD, ACT_NONE);
    k_bn_stats<<<CDIM, 256, 0, stream>>>(outb, N_NODES, mu, var);
    k_bn_res_softplus<<<divup(NC, TB), TB, 0, stream>>>(h, outb, mu, var, ebng, ebnb, NC);

    run_att(att[1]);
    run_att(att[2]);

    // --- graph pooling + head ---
    hipMemsetAsync(gpool, 0, (size_t)NGRAPH * CDIM * sizeof(float), stream);
    hipMemsetAsync(gcnt,  0, (size_t)NGRAPH * sizeof(float), stream);
    k_scatter_add<<<divup(NC, TB), TB, 0, stream>>>(h, batch, gpool, NC, CDIM);
    k_counts<<<divup(N_NODES, TB), TB, 0, stream>>>(batch, gcnt, N_NODES);
    k_divrows<<<divup(NGRAPH * CDIM, TB), TB, 0, stream>>>(gpool, gcnt, NGRAPH * CDIM, CDIM);
    gemm(gpool, Pfc, fcb, g2, NGRAPH, CDIM, CDIM, CDIM, ACT_SILU);
    gemm(g2, Pfo, fob, (float*)d_out, NGRAPH, 1, CDIM, CDIM, ACT_NONE);
}
